// GAttention_35184372089099
// MI455X (gfx1250) — compile-verified, matrix operand order checked by reading, not run on hardware
//
#include <hip/hip_runtime.h>

#define S_LEN  2048
#define EMB    64
#define HEADS  12
#define HD     128
#define INNER  1536   // HEADS*HD

typedef __attribute__((ext_vector_type(16))) __bf16 v16bf;
typedef __attribute__((ext_vector_type(8)))  __bf16 v8bf;
typedef __attribute__((ext_vector_type(8)))  float  v8f;

// Load a 16x32 bf16 fragment where each lane's 16 values are contiguous in
// memory (A-matrix: lane = row; B-matrix: lane = column of B when B's column
// is contiguous in memory).
// CDNA5 16-bit layout: lanes 0-15 kb=0 (k 0..7,16..23), lanes 16-31 kb=8.
static __device__ __forceinline__ v16bf frag_contig(const __bf16* p, int ld, int lane) {
  int r  = lane & 15;
  int kb = (lane >> 4) * 8;
  const __bf16* q = p + (size_t)r * ld + kb;
  v8bf lo = *(const v8bf*)(q);
  v8bf hi = *(const v8bf*)(q + 16);
  v16bf out;
#pragma unroll
  for (int i = 0; i < 8; ++i) { out[i] = lo[i]; out[8 + i] = hi[i]; }
  return out;
}

// ---------------------------------------------------------------------------
// Kernel 1: fused QKV projection + RoPE, f32 in -> bf16 out.
//   Q,K -> [B,H,S,D] (row-major per head),  V -> TRANSPOSED [B,H,D,S] so the
//   attention kernel can load V B-fragments as contiguous 16-byte pieces.
// grid = (256 Mtiles, 96 Ntiles, 3 matrices), block = 1 wave
// ---------------------------------------------------------------------------
__global__ __launch_bounds__(32) void k_proj_rope(
    const float* __restrict__ x,
    const float* __restrict__ Wq, const float* __restrict__ Wk,
    const float* __restrict__ Wv,
    __bf16* __restrict__ Qh, __bf16* __restrict__ Kh, __bf16* __restrict__ Vt) {
  int lane  = threadIdx.x;
  int m0    = blockIdx.x * 16;
  int n0    = blockIdx.y * 16;
  int which = blockIdx.z;
  const float* W   = (which == 0) ? Wq : (which == 1) ? Wk : Wv;
  __bf16*      dst = (which == 0) ? Qh : (which == 1) ? Kh : Vt;

  int r  = lane & 15;
  int hi = lane >> 4;
  int kb = hi * 8;

  // A: rows of x (lane = row), B: rows of W = columns of W^T (lane = out chan)
  const float* xp = x + (size_t)(m0 + r) * EMB;
  const float* wp = W + (size_t)(n0 + r) * EMB;
  v16bf a0, a1, b0, b1;
#pragma unroll
  for (int i = 0; i < 8; ++i) {
    a0[i]     = (__bf16)xp[kb + i];
    a0[8 + i] = (__bf16)xp[16 + kb + i];
    a1[i]     = (__bf16)xp[32 + kb + i];
    a1[8 + i] = (__bf16)xp[48 + kb + i];
    b0[i]     = (__bf16)wp[kb + i];
    b0[8 + i] = (__bf16)wp[16 + kb + i];
    b1[i]     = (__bf16)wp[32 + kb + i];
    b1[8 + i] = (__bf16)wp[48 + kb + i];
  }
  v8f c = {};
  c = __builtin_amdgcn_wmma_f32_16x16x32_bf16(false, a0, false, b0, (short)0, c, false, false);
  c = __builtin_amdgcn_wmma_f32_16x16x32_bf16(false, a1, false, b1, (short)0, c, false, false);

  int n = n0 + r;           // global channel in [0,1536)
  int h = n >> 7;           // head
  int d = n & 127;          // dim within head
  float inv_freq = __expf(-(float)(d >> 1) * (9.210340371976184f / 64.0f));

#pragma unroll
  for (int j = 0; j < 8; ++j) {
    int   m = m0 + j + 8 * hi;     // global row in [0,4096)
    float v = c[j];
    int   b = m >> 11, s = m & (S_LEN - 1);
    if (which < 2) {               // RoPE on Q and K
      float vp = __shfl_xor(v, 1, 32);  // pair partner = adjacent channel
      float ang = (float)s * inv_freq;
      float sn, cs;
      __sincosf(ang, &sn, &cs);
      v = ((d & 1) == 0) ? (v * cs - vp * sn) : (vp * sn + v * cs);
      dst[(((size_t)b * HEADS + h) * S_LEN + s) * HD + d] = (__bf16)v;
    } else {                       // V stored transposed: [B,H,D,S]
      dst[(((size_t)b * HEADS + h) * HD + d) * S_LEN + s] = (__bf16)v;
    }
  }
}

// ---------------------------------------------------------------------------
// Kernel 2: causal flash attention.  1 wave = 16 queries, 32-key chunks.
// grid = (S/16/4, HEADS, B), block = 128 (4 independent waves)
// ---------------------------------------------------------------------------
__global__ __launch_bounds__(128) void k_attn(
    const __bf16* __restrict__ Qh, const __bf16* __restrict__ Kh,
    const __bf16* __restrict__ Vt, __bf16* __restrict__ Oh) {
  __shared__ alignas(16) __bf16 Pl[4][16 * 32];  // per-wave P staging

  int lane = threadIdx.x & 31;
  int wave = threadIdx.x >> 5;
  int mt   = blockIdx.x * 4 + wave;
  int h    = blockIdx.y, b = blockIdx.z;

  const __bf16* Qp  = Qh + ((size_t)b * HEADS + h) * S_LEN * HD;
  const __bf16* Kp  = Kh + ((size_t)b * HEADS + h) * S_LEN * HD;
  const __bf16* Vtp = Vt + ((size_t)b * HEADS + h) * HD * S_LEN;  // [D][S]

  int q0  = mt * 16;
  int r   = lane & 15;
  int hi  = lane >> 4;
  int kb  = hi * 8;
  __bf16* P = Pl[wave];

  // Q fragments: 4 K-slices of 32 over D=128
  v16bf qf[4];
#pragma unroll
  for (int t = 0; t < 4; ++t)
    qf[t] = frag_contig(Qp + (size_t)q0 * HD + t * 32, HD, lane);

  v8f   o[8];
  float rmax[8], rsum[8];
#pragma unroll
  for (int j = 0; j < 8; ++j) { o[j] = (v8f){}; rmax[j] = -1e30f; rsum[j] = 0.0f; }

  const float scale = 0.08838834764831845f;  // 1/sqrt(128)

  for (int kc = 0; kc <= q0 + 15; kc += 32) {
    // prefetch next chunk of K (row-major) and V^T columns into caches
    if (kc + 32 <= q0 + 15) {
      __builtin_prefetch(Kp + (size_t)(kc + 32) * HD + lane * 8, 0, 0);
      __builtin_prefetch(Vtp + (size_t)lane * S_LEN + kc + 32, 0, 0);
    }

    // ---- scores: S[16 x 32] = Q(16x128) x K^T, two 16-col tiles, K=128
    v8f s0 = {}, s1 = {};
#pragma unroll
    for (int t = 0; t < 4; ++t) {
      v16bf kf0 = frag_contig(Kp + (size_t)kc * HD + t * 32, HD, lane);
      v16bf kf1 = frag_contig(Kp + (size_t)(kc + 16) * HD + t * 32, HD, lane);
      s0 = __builtin_amdgcn_wmma_f32_16x16x32_bf16(false, qf[t], false, kf0, (short)0, s0, false, false);
      s1 = __builtin_amdgcn_wmma_f32_16x16x32_bf16(false, qf[t], false, kf1, (short)0, s1, false, false);
    }
    bool need_mask = (kc + 31 > q0);
#pragma unroll
    for (int j = 0; j < 8; ++j) {
      int   row = q0 + j + 8 * hi;
      float x0 = s0[j] * scale, x1 = s1[j] * scale;
      if (need_mask) {
        if (kc + r      > row) x0 = -1e30f;
        if (kc + 16 + r > row) x1 = -1e30f;
      }
      s0[j] = x0; s1[j] = x1;
    }
    // ---- online softmax; rows live in 16-lane halves of the C layout
    float ml[8];
#pragma unroll
    for (int j = 0; j < 8; ++j) ml[j] = fmaxf(s0[j], s1[j]);
#pragma unroll
    for (int msk = 1; msk < 16; msk <<= 1)
#pragma unroll
      for (int j = 0; j < 8; ++j) ml[j] = fmaxf(ml[j], __shfl_xor(ml[j], msk, 32));

    float corr[8], rs[8];
#pragma unroll
    for (int j = 0; j < 8; ++j) {
      float mn = fmaxf(rmax[j], ml[j]);
      corr[j]  = __expf(rmax[j] - mn);
      rmax[j]  = mn;
      s0[j]    = __expf(s0[j] - mn);
      s1[j]    = __expf(s1[j] - mn);
      rs[j]    = s0[j] + s1[j];
    }
#pragma unroll
    for (int msk = 1; msk < 16; msk <<= 1)
#pragma unroll
      for (int j = 0; j < 8; ++j) rs[j] += __shfl_xor(rs[j], msk, 32);
#pragma unroll
    for (int j = 0; j < 8; ++j) rsum[j] = rsum[j] * corr[j] + rs[j];
#pragma unroll
    for (int t = 0; t < 8; ++t)
#pragma unroll
      for (int j = 0; j < 8; ++j) o[t][j] *= corr[j];

    // ---- C-layout -> A-layout for P via per-wave LDS tile (in-order DS pipe)
#pragma unroll
    for (int j = 0; j < 8; ++j) {
      P[(j + 8 * hi) * 32 + r]      = (__bf16)s0[j];
      P[(j + 8 * hi) * 32 + 16 + r] = (__bf16)s1[j];
    }
    asm volatile("" ::: "memory");
    v8bf plo = *(const v8bf*)&P[r * 32 + kb];
    v8bf phi = *(const v8bf*)&P[r * 32 + 16 + kb];
    v16bf pf;
#pragma unroll
    for (int i = 0; i < 8; ++i) { pf[i] = plo[i]; pf[8 + i] = phi[i]; }

    // ---- O += P(16x32) x V(32x128): 8 D-tiles; V^T columns are contiguous
#pragma unroll
    for (int t = 0; t < 8; ++t) {
      v16bf vf = frag_contig(Vtp + (size_t)(t * 16) * S_LEN + kc, S_LEN, lane);
      o[t] = __builtin_amdgcn_wmma_f32_16x16x32_bf16(false, pf, false, vf, (short)0, o[t], false, false);
    }
  }

  // ---- normalize and store O as bf16 in [B, S, H*D]
  float inv[8];
#pragma unroll
  for (int j = 0; j < 8; ++j) inv[j] = 1.0f / rsum[j];
#pragma unroll
  for (int t = 0; t < 8; ++t)
#pragma unroll
    for (int j = 0; j < 8; ++j) {
      int row = q0 + j + 8 * hi;
      Oh[((size_t)b * S_LEN + row) * INNER + h * HD + t * 16 + r] =
          (__bf16)(o[t][j] * inv[j]);
    }
}

// ---------------------------------------------------------------------------
// Kernel 3: output projection out = O @ Wo^T  ([4096,1536] x [1536,64])
// grid = (256 Mtiles, 4 Ntiles), block = 1 wave, K loop = 48 WMMAs
// ---------------------------------------------------------------------------
__global__ __launch_bounds__(32) void k_outproj(
    const __bf16* __restrict__ Oh, const float* __restrict__ Wo,
    float* __restrict__ out) {
  int lane = threadIdx.x;
  int m0 = blockIdx.x * 16, n0 = blockIdx.y * 16;
  int r = lane & 15, hi = lane >> 4, kb = hi * 8;

  const float* wp = Wo + (size_t)(n0 + r) * INNER;
  v8f c = {};
#pragma unroll 2
  for (int ks = 0; ks < INNER; ks += 32) {
    v16bf a = frag_contig(Oh + (size_t)m0 * INNER + ks, INNER, lane);
    v16bf bf;
#pragma unroll
    for (int i = 0; i < 8; ++i) {
      bf[i]     = (__bf16)wp[ks + kb + i];
      bf[8 + i] = (__bf16)wp[ks + 16 + kb + i];
    }
    c = __builtin_amdgcn_wmma_f32_16x16x32_bf16(false, a, false, bf, (short)0, c, false, false);
  }
#pragma unroll
  for (int j = 0; j < 8; ++j)
    out[(size_t)(m0 + j + 8 * hi) * EMB + n0 + r] = c[j];
}

// ---------------------------------------------------------------------------
extern "C" void kernel_launch(void* const* d_in, const int* in_sizes, int n_in,
                              void* d_out, int out_size, void* d_ws, size_t ws_size,
                              hipStream_t stream) {
  const float* q  = (const float*)d_in[0];
  const float* Wq = (const float*)d_in[1];
  const float* Wk = (const float*)d_in[2];
  const float* Wv = (const float*)d_in[3];
  const float* Wo = (const float*)d_in[4];
  float* out = (float*)d_out;

  // bf16 intermediates: Qh/Kh [B,H,S,D], Vt [B,H,D,S], Oh [B,S,H*D];
  // each plane 4096*1536 = 6,291,456 elems -> 4 * 12 MB = 48 MB total,
  // all resident in the 192 MB L2 across kernels.
  __bf16* ws = (__bf16*)d_ws;
  const size_t plane = (size_t)2 * HEADS * S_LEN * HD;
  __bf16* Qh = ws;
  __bf16* Kh = ws + plane;
  __bf16* Vt = ws + 2 * plane;
  __bf16* Oh = ws + 3 * plane;

  dim3 g1(256, 96, 3);          // (M/16, INNER/16, {Q,K,V})
  k_proj_rope<<<g1, 32, 0, stream>>>(q, Wq, Wk, Wv, Qh, Kh, Vt);

  dim3 g2(S_LEN / 16 / 4, HEADS, 2);
  k_attn<<<g2, 128, 0, stream>>>(Qh, Kh, Vt, Oh);

  dim3 g3(256, 4, 1);           // (M/16, EMB/16)
  k_outproj<<<g3, 32, 0, stream>>>(Oh, Wo, out);
}
